// HeteroGraphSAGE_2388001816836
// MI455X (gfx1250) — matrix operand order repaired and checked by reading, not compiled, not run on hardware
//
#include <hip/hip_runtime.h>

// ---------------------------------------------------------------------------
// HeteroGraphSAGE forward for MI455X (gfx1250, wave32).
//  - scatter-mean: edge-parallel, global_atomic_add_f32 (memory-bound part)
//  - SAGE tile kernel: fused mean-div + f16 convert + WMMA GEMM over K=256
//    ([agg | x] vs stacked [Wl ; Wr]) + bias + row L2-norm + accumulate
//  - layer-2 computed only for the 'application' destination (mer/dev dead)
// ---------------------------------------------------------------------------

typedef _Float16 half16 __attribute__((ext_vector_type(16)));
typedef _Float16 half8  __attribute__((ext_vector_type(8)));
typedef float    float8 __attribute__((ext_vector_type(8)));

#define DFEAT 128          // input feature dim
#define HFEAT 128          // hidden dim
#define KTOT  256          // concat [agg(128) | x(128)]

// ---------------------------------------------------------------- utilities
__global__ void zero_f32(float* __restrict__ p, long long n) {
  long long i = (long long)blockIdx.x * blockDim.x + threadIdx.x;
  long long stride = (long long)gridDim.x * blockDim.x;
  for (; i < n; i += stride) p[i] = 0.0f;
}

__global__ void relu_f32(float* __restrict__ p, long long n) {
  long long i = (long long)blockIdx.x * blockDim.x + threadIdx.x;
  long long stride = (long long)gridDim.x * blockDim.x;
  for (; i < n; i += stride) p[i] = fmaxf(p[i], 0.0f);
}

// Pack f32 Wl[4][128][128], Wr[4][128][128] into f16 Wc[4*128][256] rows:
// Wc[k*128+n][0:128] = Wl[k][n][:], Wc[k*128+n][128:256] = Wr[k][n][:]
__global__ void cvt_weights(const float* __restrict__ Wl,
                            const float* __restrict__ Wr,
                            _Float16* __restrict__ Wc) {
  int idx = blockIdx.x * blockDim.x + threadIdx.x;   // over 4*128*128
  if (idx >= 4 * 128 * 128) return;
  int dcol = idx & 127;
  int row  = idx >> 7;                               // k*128 + n
  Wc[(long long)row * KTOT + dcol]         = (_Float16)Wl[idx];
  Wc[(long long)row * KTOT + DFEAT + dcol] = (_Float16)Wr[idx];
}

// ------------------------------------------------------------- scatter-mean
// One edge per 32-lane group; each lane gathers float4 of the source row and
// atomically adds into the destination accumulator. Lane 0 bumps the count.
__global__ void scatter_sum(const float* __restrict__ xsrc,
                            const int* __restrict__ eidx,   // [2][E] row-major
                            float* __restrict__ agg,
                            float* __restrict__ cnt,
                            int nE) {
  int gid  = blockIdx.x * blockDim.x + threadIdx.x;
  int e    = gid >> 5;
  int lane = gid & 31;
  if (e >= nE) return;
  int src = eidx[e];
  int dst = eidx[nE + e];
  const float4 v = *(const float4*)(xsrc + (long long)src * DFEAT + lane * 4);
  float* a = agg + (long long)dst * DFEAT + lane * 4;
  atomicAdd(a + 0, v.x);
  atomicAdd(a + 1, v.y);
  atomicAdd(a + 2, v.z);
  atomicAdd(a + 3, v.w);
  if (lane == 0) atomicAdd(cnt + dst, 1.0f);
}

// ----------------------------------------------------------- fused SAGE tile
// Block = 256 threads = 8 waves. Block handles 16 dst nodes; wave w produces
// output columns [16w, 16w+16). c = (agg/cnt)@Wl^T + bias + x@Wr^T expressed
// as a single K=256 GEMM against Wc; then row L2-norm; then out += c/||c||.
__global__ __launch_bounds__(256)
void sage_tile(const float* __restrict__ agg,
               const float* __restrict__ cnt,
               const float* __restrict__ xd,
               const _Float16* __restrict__ Wc,   // [128][256] (this edge type)
               const float* __restrict__ bias,    // [128]
               float* __restrict__ out,           // [Nd][128], accumulated
               int Nd) {
  __shared__ __align__(32) _Float16 As[16][KTOT];  // 8 KB
  __shared__ float Cbuf[16][HFEAT];                // 8 KB
  __shared__ float partial[16][16];
  __shared__ float norms[16];

  const int tid  = threadIdx.x;
  const int wave = tid >> 5;
  const int lane = tid & 31;
  const int nodeBase = blockIdx.x * 16;

  // ---- stage A = [agg/cnt | x] as f16 in LDS: thread t -> row t/16, 8 cols
  {
    const int row = tid >> 4;
    const int c0  = (tid & 15) * 8;
    const int node = nodeBase + row;
    if (node < Nd) {
      const float inv = 1.0f / fmaxf(cnt[node], 1.0f);
      const float* ap = agg + (long long)node * DFEAT + c0;
      const float* xp = xd  + (long long)node * DFEAT + c0;
#pragma unroll
      for (int i = 0; i < 8; ++i) {
        As[row][c0 + i]         = (_Float16)(ap[i] * inv);
        As[row][DFEAT + c0 + i] = (_Float16)xp[i];
      }
    } else {
#pragma unroll
      for (int i = 0; i < 8; ++i) {
        As[row][c0 + i]         = (_Float16)0.0f;
        As[row][DFEAT + c0 + i] = (_Float16)0.0f;
      }
    }
  }
  __syncthreads();

  // ---- WMMA: 8 K-tiles of 32; full blocks -> EXEC all ones
  const int g   = lane >> 4;        // K half-group
  const int lm  = lane & 15;        // A row / B,C column within tile
  const int col = wave * 16 + lm;   // global output column
  float8 acc = {};
#pragma unroll
  for (int kt = 0; kt < 8; ++kt) {
    // A frag: lane holds row lm, K runs [16g,16g+16)
    half8 alo = *(const half8*)&As[lm][kt * 32 + 16 * g];
    half8 ahi = *(const half8*)&As[lm][kt * 32 + 16 * g + 8];
    half16 aF = __builtin_shufflevector(alo, ahi,
        0, 1, 2, 3, 4, 5, 6, 7, 8, 9, 10, 11, 12, 13, 14, 15);
    // B frag: lane holds column `col`, K = 16g + e -> 16 contiguous halves
    half16 bF = *(const half16*)(Wc + (long long)col * KTOT + kt * 32 + 16 * g);
    acc = __builtin_amdgcn_wmma_f32_16x16x32_f16(
        /*neg_a=*/false, aF, /*neg_b=*/false, bF,
        /*c_mod=*/(short)0, acc, /*reuse_a=*/false, /*reuse_b=*/false);
  }

  // ---- bias + spill C tile to LDS (row = r + 8g, col = col)
  {
    const float b = bias[col];
#pragma unroll
    for (int r = 0; r < 8; ++r)
      Cbuf[r + 8 * g][col] = acc[r] + b;
  }
  __syncthreads();

  // ---- row-wise sum of squares: 16 rows x 16 segments of 8
  {
    const int r = tid >> 4, s0 = (tid & 15) * 8;
    float s = 0.0f;
#pragma unroll
    for (int i = 0; i < 8; ++i) { float v = Cbuf[r][s0 + i]; s += v * v; }
    partial[r][tid & 15] = s;
  }
  __syncthreads();
  if (tid < 16) {
    float s = 0.0f;
#pragma unroll
    for (int i = 0; i < 16; ++i) s += partial[tid][i];
    norms[tid] = fmaxf(sqrtf(s), 1e-12f);
  }
  __syncthreads();

  // ---- out[node] += c / ||c||  (block exclusively owns these 16 nodes)
  {
    const int r = tid >> 4, c0 = (tid & 15) * 8;
    const int node = nodeBase + r;
    if (node < Nd) {
      const float inv = 1.0f / norms[r];
      float* op = out + (long long)node * HFEAT + c0;
#pragma unroll
      for (int i = 0; i < 8; ++i) op[i] += Cbuf[r][c0 + i] * inv;
    }
  }
}

// ------------------------------------------------------------- classifier
__global__ void classifier(const float* __restrict__ h,   // [N][128]
                           const float* __restrict__ W,   // [2][128]
                           const float* __restrict__ b,   // [2]
                           float* __restrict__ out,       // [N][2]
                           int N) {
  int node = blockIdx.x * blockDim.x + threadIdx.x;
  if (node >= N) return;
  float a0 = b[0], a1 = b[1];
  const float* hp = h + (long long)node * HFEAT;
#pragma unroll 4
  for (int j = 0; j < HFEAT; ++j) {
    float v = hp[j];
    a0 += v * W[j];
    a1 += v * W[HFEAT + j];
  }
  out[node * 2 + 0] = a0;
  out[node * 2 + 1] = a1;
}

// ---------------------------------------------------------------------------
extern "C" void kernel_launch(void* const* d_in, const int* in_sizes, int n_in,
                              void* d_out, int out_size, void* d_ws, size_t ws_size,
                              hipStream_t stream) {
  (void)n_in; (void)out_size; (void)ws_size;

  const float* emb_app  = (const float*)d_in[0];
  const float* emb_mer  = (const float*)d_in[1];
  const float* emb_dev  = (const float*)d_in[2];
  const float* conv1_Wl = (const float*)d_in[3];
  const float* conv1_bl = (const float*)d_in[4];
  const float* conv1_Wr = (const float*)d_in[5];
  const float* conv2_Wl = (const float*)d_in[6];
  const float* conv2_bl = (const float*)d_in[7];
  const float* conv2_Wr = (const float*)d_in[8];
  const float* clf_W    = (const float*)d_in[9];
  const float* clf_b    = (const float*)d_in[10];
  const int* eidx[4] = { (const int*)d_in[11], (const int*)d_in[12],
                         (const int*)d_in[13], (const int*)d_in[14] };
  const int E = in_sizes[11] / 2;

  const int N_APP = 200000, N_MER = 50000, N_DEV = 100000;
  const int sizes[3] = { N_APP, N_MER, N_DEV };
  const int ets_s[4] = { 0, 1, 0, 2 };
  const int ets_d[4] = { 1, 0, 2, 0 };

  // -------- workspace carve-out
  char* ws = (char*)d_ws;
  size_t off = 0;
  auto carve = [&](size_t bytes) -> void* {
    size_t o = (off + 255) & ~(size_t)255;
    off = o + bytes;
    return (void*)(ws + o);
  };
  float* h1_app = (float*)carve((size_t)N_APP * HFEAT * 4);
  float* h1_mer = (float*)carve((size_t)N_MER * HFEAT * 4);
  float* h1_dev = (float*)carve((size_t)N_DEV * HFEAT * 4);
  float* h2_app = (float*)carve((size_t)N_APP * HFEAT * 4);
  float* aggb   = (float*)carve((size_t)N_APP * DFEAT * 4);   // max dst = app
  float* cntb   = (float*)carve((size_t)N_APP * 4);
  _Float16* Wc1 = (_Float16*)carve((size_t)4 * 128 * KTOT * 2);
  _Float16* Wc2 = (_Float16*)carve((size_t)4 * 128 * KTOT * 2);

  const float* xs1[3] = { emb_app, emb_mer, emb_dev };
  const float* xs2[3] = { h1_app,  h1_mer,  h1_dev  };
  float*       h1[3]  = { h1_app,  h1_mer,  h1_dev  };

  auto zero = [&](float* p, long long n) {
    zero_f32<<<2048, 256, 0, stream>>>(p, n);
  };

  // -------- weight packing (f32 -> f16, [Wl | Wr] per row)
  {
    int nthr = 4 * 128 * 128;
    cvt_weights<<<(nthr + 255) / 256, 256, 0, stream>>>(conv1_Wl, conv1_Wr, Wc1);
    cvt_weights<<<(nthr + 255) / 256, 256, 0, stream>>>(conv2_Wl, conv2_Wr, Wc2);
  }

  // -------- layer 1: all 4 edge types, accumulate into h1[dst]
  zero(h1_app, (long long)N_APP * HFEAT);
  zero(h1_mer, (long long)N_MER * HFEAT);
  zero(h1_dev, (long long)N_DEV * HFEAT);
  const int scatterBlocks = (E * 32 + 255) / 256;
  for (int k = 0; k < 4; ++k) {
    const int s = ets_s[k], d = ets_d[k];
    const int Nd = sizes[d];
    zero(aggb, (long long)Nd * DFEAT);
    zero(cntb, Nd);
    scatter_sum<<<scatterBlocks, 256, 0, stream>>>(xs1[s], eidx[k], aggb, cntb, E);
    sage_tile<<<(Nd + 15) / 16, 256, 0, stream>>>(
        aggb, cntb, xs1[d], Wc1 + (size_t)k * 128 * KTOT,
        conv1_bl + k * HFEAT, h1[d], Nd);
  }
  relu_f32<<<2048, 256, 0, stream>>>(h1_app, (long long)N_APP * HFEAT);
  relu_f32<<<2048, 256, 0, stream>>>(h1_mer, (long long)N_MER * HFEAT);
  relu_f32<<<2048, 256, 0, stream>>>(h1_dev, (long long)N_DEV * HFEAT);

  // -------- layer 2: only dst == application (k = 1: mer->app, k = 3: dev->app)
  zero(h2_app, (long long)N_APP * HFEAT);
  {
    const int ks[2] = { 1, 3 };
    for (int i = 0; i < 2; ++i) {
      const int k = ks[i];
      const int s = ets_s[k];
      const int Nd = N_APP;
      zero(aggb, (long long)Nd * DFEAT);
      zero(cntb, Nd);
      scatter_sum<<<scatterBlocks, 256, 0, stream>>>(xs2[s], eidx[k], aggb, cntb, E);
      sage_tile<<<(Nd + 15) / 16, 256, 0, stream>>>(
          aggb, cntb, h1_app, Wc2 + (size_t)k * 128 * KTOT,
          conv2_bl + k * HFEAT, h2_app, Nd);
    }
  }

  // -------- classifier: logits for application nodes
  classifier<<<(N_APP + 255) / 256, 256, 0, stream>>>(
      h2_app, clf_W, clf_b, (float*)d_out, N_APP);
}
